// ConvolutionalLayer_83751862272232
// MI455X (gfx1250) — matrix operand order repaired and checked by reading, not compiled, not run on hardware
//
#include <hip/hip_runtime.h>

typedef __attribute__((ext_vector_type(16))) _Float16 v16h;
typedef __attribute__((ext_vector_type(8)))  _Float16 v8h;
typedef __attribute__((ext_vector_type(8)))  float    v8f;
typedef __attribute__((ext_vector_type(4)))  unsigned v4u;
typedef __attribute__((ext_vector_type(8)))  int      v8i;
typedef __attribute__((ext_vector_type(4)))  int      v4i;

#define H_    512
#define W_    512
#define CIN   64
#define COUT  128
#define P_    516            // zero-padded H/W for the f16 input copy
#define BPX   128            // pixels per block (one row segment)
#define HALO  (BPX + 4)      // 132 columns of halo
#define THRESH 15.0f

#define XP_HALVES ((size_t)P_ * P_ * CIN)   // padded f16 x: 17,040,384 halves
#define BP_HALVES ((size_t)50 * 8 * 32 * 16) // packed weights: 204,800 halves

// ---------------------------------------------------------------------------
// Prepass 1: zero-padded f16 copy of x: xp[py][px][ci], py/px in [0,516).
// ---------------------------------------------------------------------------
__global__ __launch_bounds__(256) void pad_convert(const float* __restrict__ x,
                                                   _Float16* __restrict__ xp) {
  size_t i = (size_t)blockIdx.x * 256 + threadIdx.x;
  if (i >= XP_HALVES) return;
  int ci = (int)(i & (CIN - 1));
  size_t p = i >> 6;
  int px = (int)(p % P_);
  int py = (int)(p / P_);
  int gx = px - 2, gy = py - 2;
  float v = 0.0f;
  if ((unsigned)gx < (unsigned)W_ && (unsigned)gy < (unsigned)H_)
    v = x[((size_t)(gy * W_ + gx) << 6) | ci];
  xp[i] = (_Float16)v;
}

// ---------------------------------------------------------------------------
// Prepass 2: pack f32 HWIO weights into f16 WMMA B fragments.
// Bp[chunk(50)][ntile(8)][lane(32)][elem(16)];  chunk = tap*2 + ci-half,
// B element (k,n): lane = n + 16*(k>=16), elem = k%16  -> 32B/lane fragment.
// ---------------------------------------------------------------------------
__global__ __launch_bounds__(256) void pack_weights(const float* __restrict__ wsrc,
                                                    _Float16* __restrict__ bp) {
  int i = blockIdx.x * 256 + threadIdx.x;
  if (i >= (int)BP_HALVES) return;
  int j     = i & 15;
  int lane  = (i >> 4) & 31;
  int nt    = (i >> 9) & 7;
  int chunk = i >> 12;
  int tap   = chunk >> 1;
  int h     = chunk & 1;
  int ci    = h * 32 + ((lane >> 4) << 4) + j;
  int n     = nt * 16 + (lane & 15);
  bp[i] = (_Float16)wsrc[((size_t)tap * CIN + ci) * COUT + n];
}

// ---------------------------------------------------------------------------
// Main kernel: block = 1 row x 128 pixels x 128 channels; 8 waves =
// 2 M-groups (64 px each) x 4 N-groups (32 ch each). Each wave: 4 M-tiles,
// 2 N-tiles -> per K-chunk: 4 A loads + 2 B loads feed 8 WMMAs.
// TDM=true: halo staged by one tensor_load_to_lds from the padded f16 copy.
// TDM=false: classic f32 load + convert staging (no ws for xp needed).
// ---------------------------------------------------------------------------
template <bool TDM>
__global__ __launch_bounds__(256) void snn_conv(const float* __restrict__ x,
                                                const _Float16* __restrict__ xp,
                                                const _Float16* __restrict__ bp,
                                                const float* __restrict__ oldp,
                                                const float* __restrict__ kinh,
                                                float* __restrict__ out) {
  // single shared object so its LDS offset is 0 (needed by the D# lds_addr)
  __shared__ __align__(128) unsigned char smem[5 * HALO * CIN * 2 + BPX * 4];
  _Float16* sA   = (_Float16*)smem;                       // 84480 bytes
  unsigned* sAny = (unsigned*)(smem + 5 * HALO * CIN * 2);

  const int tid  = threadIdx.x;
  const int lane = tid & 31;
  const int wv   = tid >> 5;
  const int mg   = wv >> 2;           // 0..1  : pixel half (64 px)
  const int ng   = wv & 3;            // 0..3  : channel group (32 ch)
  const int x0   = blockIdx.x * BPX;
  const int y    = blockIdx.y;

  for (int i = tid; i < BPX; i += 256) sAny[i] = 0u;

  if constexpr (TDM) {
    if (wv == 0) {
      // ---- Tensor DMA: 2D tile, 5 rows x (132*64) f16, stride 516*64 ----
      constexpr unsigned TD0   = (unsigned)P_ * CIN;   // 33024 elems / row
      constexpr unsigned TILE0 = (unsigned)HALO * CIN; // 8448 elems / tile row
      unsigned long long ga = (unsigned long long)(uintptr_t)xp +
                              (((size_t)y * P_ + x0) * CIN) * 2ull;
      v4u g0 = { 1u,                                   // count=1, load, normal
                 0u,                                   // lds_addr = 0 (sA)
                 (unsigned)(ga & 0xFFFFFFFFu),
                 (unsigned)((ga >> 32) & 0x1FFFFFFu) | (2u << 30) }; // type=2
      v8i g1 = { (int)(1u << 16),                      // data_size = 2 bytes
                 (int)((TD0 & 0xFFFFu) << 16),         // tensor_dim0 lo16
                 (int)(((TD0 >> 16) & 0xFFFFu) | ((unsigned)P_ << 16)),
                 (int)((TILE0 << 16)),                 // tile_dim0 (dim1 hi16=0)
                 (int)5,                               // tile_dim1 = 5 rows
                 (int)TD0,                             // dim0_stride lo32
                 0, 0 };
      v4i gz = { 0, 0, 0, 0 };
#if __clang_major__ >= 23
      v8i gz8 = { 0, 0, 0, 0, 0, 0, 0, 0 };
      __builtin_amdgcn_tensor_load_to_lds(g0, g1, gz, gz, gz8, 0);
#else
      __builtin_amdgcn_tensor_load_to_lds(g0, g1, gz, gz, 0);
#endif
      __builtin_amdgcn_s_wait_tensorcnt(0);
    }
  } else {
    for (int idx = tid; idx < 5 * HALO * CIN; idx += 256) {
      int ci  = idx & 63;
      int col = (idx >> 6) % HALO;
      int r   = idx / (HALO * CIN);
      int gy  = y + r - 2;
      int gx  = x0 + col - 2;
      float v = 0.0f;
      if ((unsigned)gy < (unsigned)H_ && (unsigned)gx < (unsigned)W_)
        v = x[((size_t)(gy * W_ + gx) << 6) | ci];
      sA[idx] = (_Float16)v;
    }
  }
  __syncthreads();

  const int m      = lane & 15;        // A: M = lane%16
  const int hiHalf = lane >> 4;
  const int aBase0 = hiHalf * 16;      // A K pattern: lo lanes 0-15, hi 16-31

  v8f acc[2][4] = {};

  for (int tap = 0; tap < 25; ++tap) {
    const int ky   = tap / 5;
    const int kx   = tap - ky * 5;
    const int arow = ky * HALO + kx + mg * 64 + m;
    #pragma unroll
    for (int h = 0; h < 2; ++h) {
      const int chunk = tap * 2 + h;
      const _Float16* bbase = bp + (((size_t)chunk * 8) * 32 + lane) * 16;
      v16h b0 = *(const v16h*)(bbase + (size_t)(ng * 2 + 0) * 32 * 16);
      v16h b1 = *(const v16h*)(bbase + (size_t)(ng * 2 + 1) * 32 * 16);
      const int ciA = h * 32 + aBase0;
      v16h a[4];
      #pragma unroll
      for (int t = 0; t < 4; ++t) {
        const _Float16* ap = &sA[(size_t)(arow + t * 16) * CIN + ciA];
        v8h alo = *(const v8h*)ap;        // ds_load_b128
        v8h ahi = *(const v8h*)(ap + 8);  // ds_load_b128
        a[t] = __builtin_shufflevector(alo, ahi,
                 0,1,2,3,4,5,6,7,8,9,10,11,12,13,14,15);
      }
      #pragma unroll
      for (int t = 0; t < 4; ++t)
        acc[0][t] = __builtin_amdgcn_wmma_f32_16x16x32_f16(
                      false, a[t], false, b0, (short)0, acc[0][t], false, false);
      #pragma unroll
      for (int t = 0; t < 4; ++t)
        acc[1][t] = __builtin_amdgcn_wmma_f32_16x16x32_f16(
                      false, a[t], false, b1, (short)0, acc[1][t], false, false);
    }
  }

  // ---------------- epilogue: spike / reset / lateral inhibition ------------
  const size_t SP = (size_t)H_ * W_ * COUT;  // 33554432
  const int n  = lane & 15;                  // C/D: lane%16 = N
  const int mB = hiHalf * 8;                 // C/D: vgpr r -> M = r + 8*hiHalf

  #pragma unroll
  for (int t = 0; t < 4; ++t) {
    #pragma unroll
    for (int r = 0; r < 8; ++r) {
      int lpx = mg * 64 + t * 16 + mB + r;   // local pixel 0..127
      size_t pidx = (size_t)y * W_ + (size_t)(x0 + lpx);
      float kv = kinh[pidx];
      bool anyS = false;
      #pragma unroll
      for (int s = 0; s < 2; ++s) {
        int c = (ng * 2 + s) * 16 + n;
        float pot = acc[s][t][r] + oldp[pidx * COUT + c];
        bool spiked = pot >= THRESH;
        anyS |= spiked;
        out[SP + pidx * COUT + c] = spiked ? 0.0f : pot;   // reset before inhibit
        out[pidx * COUT + c] = (spiked && kv > 0.0f) ? 1.0f : 0.0f;
      }
      if (anyS) atomicOr(&sAny[lpx], 1u);    // ds_or_b32
    }
  }
  __syncthreads();

  for (int i = tid; i < BPX; i += 256) {
    size_t pidx = (size_t)y * W_ + (size_t)(x0 + i);
    float kv = kinh[pidx];
    out[2 * SP + pidx] = ((sAny[i] != 0u) && kv > 0.0f) ? 0.0f : kv;
  }
}

// ---------------------------------------------------------------------------
extern "C" void kernel_launch(void* const* d_in, const int* in_sizes, int n_in,
                              void* d_out, int out_size, void* d_ws, size_t ws_size,
                              hipStream_t stream) {
  const float* x    = (const float*)d_in[0];   // [1,512,512,64]
  const float* w    = (const float*)d_in[1];   // [5,5,64,128] HWIO
  const float* oldp = (const float*)d_in[2];   // [1,512,512,128]
  const float* kinh = (const float*)d_in[3];   // [512,512]
  float* out = (float*)d_out;

  const size_t need_fast = (XP_HALVES + BP_HALVES) * 2;  // ~34.5 MB

  if (ws_size >= need_fast) {
    _Float16* xpad = (_Float16*)d_ws;
    _Float16* bpk  = (_Float16*)d_ws + XP_HALVES;
    pad_convert<<<dim3((unsigned)((XP_HALVES + 255) / 256)), dim3(256), 0, stream>>>(x, xpad);
    pack_weights<<<dim3((unsigned)((BP_HALVES + 255) / 256)), dim3(256), 0, stream>>>(w, bpk);
    snn_conv<true><<<dim3(W_ / BPX, H_), dim3(256), 0, stream>>>(x, xpad, bpk, oldp, kinh, out);
  } else {
    _Float16* bpk = (_Float16*)d_ws;
    pack_weights<<<dim3((unsigned)((BP_HALVES + 255) / 256)), dim3(256), 0, stream>>>(w, bpk);
    snn_conv<false><<<dim3(W_ / BPX, H_), dim3(256), 0, stream>>>(x, nullptr, bpk, oldp, kinh, out);
  }
}
